// MLPResNetBlock_Pro_MoE_89876485636875
// MI455X (gfx1250) — compile-verified, matrix operand order checked by reading, not run on hardware
//
#include <hip/hip_runtime.h>
#include <stdint.h>

// ---------- types ----------
typedef float        v8f   __attribute__((ext_vector_type(8)));
typedef __bf16       v16bf __attribute__((ext_vector_type(16)));
typedef unsigned int u32x4 __attribute__((ext_vector_type(4)));
typedef unsigned int u32x2 __attribute__((ext_vector_type(2)));
typedef int          i32x4 __attribute__((ext_vector_type(4)));
typedef float        f32x4 __attribute__((ext_vector_type(4)));
typedef float        f32x2 __attribute__((ext_vector_type(2)));

#define AS1 __attribute__((address_space(1)))
#define AS3 __attribute__((address_space(3)))

#define DIMC 1024
#define NB   8
#define NT   2048
#define NH   8
#define HD   128
#define KADP 512   // adapter keys (448 + 64); task keys also 512

__device__ __forceinline__ unsigned short f2bf(float f) {
    unsigned int u = __float_as_uint(f);
    unsigned int r = u + 0x7FFFu + ((u >> 16) & 1u);   // round-to-nearest-even
    return (unsigned short)(r >> 16);
}
__device__ __forceinline__ unsigned int pk2bf(float a, float b) {
    return ((unsigned int)f2bf(b) << 16) | (unsigned int)f2bf(a);
}
__device__ __forceinline__ v8f zero8() { v8f z = {}; return z; }

__device__ __forceinline__ void wait_async4() { asm volatile("s_wait_asynccnt 0x4" ::: "memory"); }
__device__ __forceinline__ void wait_async0() { asm volatile("s_wait_asynccnt 0x0" ::: "memory"); }

// A/B fragment load (16-bit WMMA layout): lane<16 holds K{kb..kb+7, kb+16..kb+23}
__device__ __forceinline__ v16bf ld_frag_lds(const unsigned short* s, int row, int kb) {
    union { u32x4 q[2]; v16bf v; } f;
    f.q[0] = *(const u32x4*)(s + row * 32 + kb);
    f.q[1] = *(const u32x4*)(s + row * 32 + kb + 16);
    return f.v;
}
__device__ __forceinline__ v16bf ld_frag_g(const unsigned short* g) {
    union { u32x4 q[2]; v16bf v; } f;
    f.q[0] = *(const u32x4*)g;
    f.q[1] = *(const u32x4*)(g + 16);
    return f.v;
}

// ---------- elementwise fp32 -> bf16 converts ----------
__global__ void cvt_bf16_kernel(const float* __restrict__ in, unsigned short* __restrict__ out)
{
    size_t i = ((size_t)blockIdx.x * blockDim.x + threadIdx.x) * 4;
    f32x4 v = *(const f32x4*)(in + i);
    u32x2 pk; pk.x = pk2bf(v.x, v.y); pk.y = pk2bf(v.z, v.w);
    *(u32x2*)(out + i) = pk;
}
__global__ void cvt_w_bf16_kernel(const float* __restrict__ Wall, const int* __restrict__ eptr,
                                  unsigned short* __restrict__ out)
{
    const float* in = Wall + (size_t)eptr[0] * DIMC * DIMC;
    size_t i = ((size_t)blockIdx.x * blockDim.x + threadIdx.x) * 4;
    f32x4 v = *(const f32x4*)(in + i);
    u32x2 pk; pk.x = pk2bf(v.x, v.y); pk.y = pk2bf(v.z, v.w);
    *(u32x2*)(out + i) = pk;
}

// ---------- pack h_adapter = concat(h_a, p), fp32 -> bf16 ----------
__global__ void pack_adapter_bf16_kernel(const float* __restrict__ ha, const float* __restrict__ p,
                                         unsigned short* __restrict__ out)
{
    size_t i = (size_t)blockIdx.x * blockDim.x + threadIdx.x;   // float4 index
    int row = (int)(i >> 8);
    int c4  = (int)(i & 255) * 4;
    int b = row >> 9, j = row & 511;
    const float* src = (j < 448) ? (ha + ((size_t)b * 448 + j) * DIMC)
                                 : (p  + ((size_t)b * 64 + (j - 448)) * DIMC);
    f32x4 v = *(const f32x4*)(src + c4);
    u32x2 pk; pk.x = pk2bf(v.x, v.y); pk.y = pk2bf(v.z, v.w);
    *(u32x2*)(out + (size_t)row * DIMC + c4) = pk;
}

// ---------- async-copy staging of one 128x32 bf16 tile pair ----------
__device__ __forceinline__ void stage_pair(const unsigned short* Ab, const unsigned short* Wb,
                                           unsigned short* As, unsigned short* Bs,
                                           int tid, int m0, int n0, int k0)
{
#pragma unroll
    for (int i = 0; i < 2; ++i) {
        int c = tid + i * 256;            // 0..511 -> 16B chunks of the 8 KiB tile
        int row = c >> 2, seg = (c & 3) * 8;
        __builtin_amdgcn_global_load_async_to_lds_b128(
            (AS1 i32x4*)(Ab + (size_t)(m0 + row) * DIMC + k0 + seg),
            (AS3 i32x4*)(As + row * 32 + seg), 0, 0);
        __builtin_amdgcn_global_load_async_to_lds_b128(
            (AS1 i32x4*)(Wb + (size_t)(n0 + row) * DIMC + k0 + seg),
            (AS3 i32x4*)(Bs + row * 32 + seg), 0, 0);
    }
}

// ---------- GEMM: C[M,1024] = A[M,1024](bf16) x W[1024,1024](bf16,[N][K]) + bias, opt ReLU ----------
__global__ __launch_bounds__(256)
void gemm_bf16_async_kernel(const unsigned short* __restrict__ Ab,
                            const unsigned short* __restrict__ Wb,
                            const float* __restrict__ ball, const int* __restrict__ eptr,
                            float* __restrict__ C, int relu)
{
    __shared__ unsigned short As[2][128 * 32];
    __shared__ unsigned short Bs[2][128 * 32];
    const float* bias = ball + (size_t)eptr[0] * DIMC;

    const int tid  = threadIdx.x;
    const int lane = tid & 31, w = tid >> 5;
    const int half = lane >> 4, l16 = lane & 15;
    const int m0 = blockIdx.y * 128, n0 = blockIdx.x * 128;
    const int mw = (w & 3) * 32, nw = (w >> 2) * 64;

    v8f acc[2][4];
    for (int i = 0; i < 2; ++i) for (int j = 0; j < 4; ++j) acc[i][j] = zero8();

    stage_pair(Ab, Wb, As[0], Bs[0], tid, m0, n0, 0);          // ASYNCcnt += 4

    for (int kt = 0; kt < 32; ++kt) {
        const int cur = kt & 1;
        if (kt < 31) {
            stage_pair(Ab, Wb, As[1 - cur], Bs[1 - cur], tid, m0, n0, (kt + 1) * 32);
            wait_async4();   // oldest 4 (current tile) complete; prefetch stays in flight
        } else {
            wait_async0();
        }
        __syncthreads();

        v16bf af[2], bf[4];
        for (int mt = 0; mt < 2; ++mt) af[mt] = ld_frag_lds(As[cur], mw + mt * 16 + l16, half * 8);
        for (int nt = 0; nt < 4; ++nt) bf[nt] = ld_frag_lds(Bs[cur], nw + nt * 16 + l16, half * 8);
        for (int mt = 0; mt < 2; ++mt)
            for (int nt = 0; nt < 4; ++nt)
                acc[mt][nt] = __builtin_amdgcn_wmma_f32_16x16x32_bf16(
                    false, af[mt], false, bf[nt], (short)0, acc[mt][nt], false, false);
        __syncthreads();
    }

    for (int nt = 0; nt < 4; ++nt) {
        int gc = n0 + nw + nt * 16 + l16;
        float bv = bias[gc];
        for (int mt = 0; mt < 2; ++mt)
            for (int j = 0; j < 8; ++j) {
                int gr = m0 + mw + mt * 16 + half * 8 + j;
                float v = acc[mt][nt][j] + bv;
                if (relu) v = fmaxf(v, 0.f);
                C[(size_t)gr * DIMC + gc] = v;
            }
    }
}

// ---------- RoPE + fp32->bf16 (Q and K buffers) ----------
__global__ void rope_cvt_kernel(const float* __restrict__ in, unsigned short* __restrict__ outb,
                                int seqlen)
{
    size_t i = (size_t)blockIdx.x * blockDim.x + threadIdx.x;   // pair index
    int row = (int)(i >> 9);
    int col = (int)(i & 511) * 2;
    int jp  = col & (HD - 1);
    float t = (float)(row % seqlen);
    const float LN1E4_64 = 0.143911568f;   // ln(10000)/64
    float a0 = t * __expf(-LN1E4_64 * (float)(jp & 63));
    float a1 = t * __expf(-LN1E4_64 * (float)((jp + 1) & 63));
    f32x2 x = *(const f32x2*)(in + (size_t)row * DIMC + col);
    float o0 = x.x * __cosf(a0) - x.y * __sinf(a0);
    float o1 = x.y * __cosf(a1) + x.x * __sinf(a1);
    *(unsigned int*)(outb + (size_t)row * DIMC + col) = pk2bf(o0, o1);
}

// ---------- V: fp32 [B*512,1024] -> bf16 transposed [b,h,d,key] ----------
__global__ void cvtT_v_kernel(const float* __restrict__ in, unsigned short* __restrict__ outb)
{
    size_t i = (size_t)blockIdx.x * blockDim.x + threadIdx.x;
    int row = (int)(i >> 10);   // b*512 + key
    int col = (int)(i & 1023);
    int b = row >> 9, key = row & 511;
    int h = col >> 7, d = col & 127;
    outb[(((size_t)(b * NH + h) * HD) + d) * KADP + key] = f2bf(in[i]);
}

// ---------- fused dual-stream flash attention (bf16 out) ----------
__global__ __launch_bounds__(128)
void attn_kernel(const unsigned short* __restrict__ qabf, const unsigned short* __restrict__ qtbf,
                 const unsigned short* __restrict__ kabf, const unsigned short* __restrict__ ktbf,
                 const unsigned short* __restrict__ vaT,  const unsigned short* __restrict__ vtT,
                 const float* __restrict__ gating, const int* __restrict__ eptr,
                 unsigned short* __restrict__ outb)
{
    __shared__ unsigned short Ps[4][16 * 32];
    const int bh = blockIdx.x;              // b*8 + h
    const int b = bh >> 3, h = bh & 7;
    const int w = threadIdx.x >> 5, lane = threadIdx.x & 31;
    const int half = lane >> 4, l16 = lane & 15;
    const int t0 = blockIdx.y * 64 + w * 16;
    const float ratio = 1.f / (1.f + __expf(-gating[eptr[0]]));
    const float scl = 0.0883883476483184f;  // 1/sqrt(128)

    v16bf qfa[4], qft[4];
    {
        size_t ro = ((size_t)(b * NT + t0 + l16)) * DIMC + h * HD;
        for (int kk = 0; kk < 4; ++kk) {
            qfa[kk] = ld_frag_g(qabf + ro + kk * 32 + half * 8);
            qft[kk] = ld_frag_g(qtbf + ro + kk * 32 + half * 8);
        }
    }

    v8f o[8];
    for (int i = 0; i < 8; ++i) o[i] = zero8();
    float m_s[8], l_s[8];
    for (int j = 0; j < 8; ++j) { m_s[j] = -1e30f; l_s[j] = 0.f; }

    for (int s = 0; s < 32; ++s) {
        const int  kb0  = s * 32;
        const bool task = (kb0 >= KADP);
        const float fac = task ? scl * ratio : scl;
        const unsigned short* kbase = task ? ktbf : kabf;
        const int kloc0 = task ? kb0 - KADP : kb0;

        v8f sa[2];
        for (int u = 0; u < 2; ++u) {
            sa[u] = zero8();
            const int key = kloc0 + u * 16 + l16;
            const unsigned short* krow = kbase + ((size_t)(b * KADP + key)) * DIMC + h * HD;
            for (int kk = 0; kk < 4; ++kk) {
                v16bf qq = task ? qft[kk] : qfa[kk];
                sa[u] = __builtin_amdgcn_wmma_f32_16x16x32_bf16(
                    false, qq, false, ld_frag_g(krow + kk * 32 + half * 8),
                    (short)0, sa[u], false, false);
            }
        }

        // online softmax over 32 new columns
        float corr[8];
        for (int j = 0; j < 8; ++j) {
            float v0 = sa[0][j] * fac, v1 = sa[1][j] * fac;
            float mx = fmaxf(v0, v1);
            for (int off = 1; off < 16; off <<= 1) mx = fmaxf(mx, __shfl_xor(mx, off, 32));
            float mn = fmaxf(m_s[j], mx);
            float c  = __expf(m_s[j] - mn);
            float p0 = __expf(v0 - mn), p1 = __expf(v1 - mn);
            float ps = p0 + p1;
            for (int off = 1; off < 16; off <<= 1) ps += __shfl_xor(ps, off, 32);
            l_s[j] = l_s[j] * c + ps;
            m_s[j] = mn; corr[j] = c;
            const int r = half * 8 + j;
            Ps[w][r * 32 + l16]      = f2bf(p0);
            Ps[w][r * 32 + 16 + l16] = f2bf(p1);
        }
        for (int dt = 0; dt < 8; ++dt)
            for (int j = 0; j < 8; ++j) o[dt][j] *= corr[j];

        asm volatile("s_wait_dscnt 0" ::: "memory");
        v16bf pf = ld_frag_lds(&Ps[w][0], l16, half * 8);

        const unsigned short* vb = task ? vtT : vaT;
        for (int dt = 0; dt < 8; ++dt) {
            const unsigned short* vrow =
                vb + (((size_t)(b * NH + h) * HD) + dt * 16 + l16) * KADP + kloc0 + half * 8;
            o[dt] = __builtin_amdgcn_wmma_f32_16x16x32_bf16(
                false, pf, false, ld_frag_g(vrow), (short)0, o[dt], false, false);
        }
    }

    for (int j = 0; j < 8; ++j) l_s[j] = 1.f / l_s[j];
    for (int dt = 0; dt < 8; ++dt)
        for (int j = 0; j < 8; ++j) {
            int t = t0 + half * 8 + j;
            outb[((size_t)(b * NT + t)) * DIMC + h * HD + dt * 16 + l16] = f2bf(o[dt][j] * l_s[j]);
        }
}

// ---------- residual + LayerNorm (bf16 out) ----------
__global__ __launch_bounds__(256)
void add_ln_kernel(const float* __restrict__ oo, const float* __restrict__ x,
                   const float* __restrict__ nw, const float* __restrict__ nb,
                   const int* __restrict__ eptr, unsigned short* __restrict__ outb)
{
    __shared__ float red[16];
    const int e = eptr[0];
    const float* wgt = nw + (size_t)e * DIMC;
    const float* bb  = nb + (size_t)e * DIMC;
    size_t row = blockIdx.x;
    const int c0 = threadIdx.x * 4;
    f32x4 a = *(const f32x4*)(oo + row * DIMC + c0);
    f32x4 bsum = *(const f32x4*)(x + row * DIMC + c0);
    float v[4] = { a.x + bsum.x, a.y + bsum.y, a.z + bsum.z, a.w + bsum.w };
    float s = v[0] + v[1] + v[2] + v[3];
    float s2 = v[0]*v[0] + v[1]*v[1] + v[2]*v[2] + v[3]*v[3];
    for (int off = 1; off < 32; off <<= 1) { s += __shfl_xor(s, off, 32); s2 += __shfl_xor(s2, off, 32); }
    int wv = threadIdx.x >> 5;
    if ((threadIdx.x & 31) == 0) { red[wv] = s; red[8 + wv] = s2; }
    __syncthreads();
    float ts = 0.f, ts2 = 0.f;
    for (int i = 0; i < 8; ++i) { ts += red[i]; ts2 += red[8 + i]; }
    float mu  = ts * (1.f / DIMC);
    float var = ts2 * (1.f / DIMC) - mu * mu;
    float rin = rsqrtf(var + 1e-5f);
    float o2[4];
    for (int i = 0; i < 4; ++i) o2[i] = (v[i] - mu) * rin * wgt[c0 + i] + bb[c0 + i];
    u32x2 pk; pk.x = pk2bf(o2[0], o2[1]); pk.y = pk2bf(o2[2], o2[3]);
    *(u32x2*)(outb + row * DIMC + c0) = pk;
}

// ---------- host launch ----------
extern "C" void kernel_launch(void* const* d_in, const int* in_sizes, int n_in,
                              void* d_out, int out_size, void* d_ws, size_t ws_size,
                              hipStream_t stream)
{
    const float* x    = (const float*)d_in[0];
    const float* h_a  = (const float*)d_in[1];
    const float* h_t  = (const float*)d_in[2];
    const float* p    = (const float*)d_in[3];
    const float* W_qa = (const float*)d_in[4];  const float* b_qa = (const float*)d_in[5];
    const float* W_ka = (const float*)d_in[6];  const float* b_ka = (const float*)d_in[7];
    const float* W_va = (const float*)d_in[8];  const float* b_va = (const float*)d_in[9];
    const float* W_qt = (const float*)d_in[10]; const float* b_qt = (const float*)d_in[11];
    const float* W_kt = (const float*)d_in[12]; const float* b_kt = (const float*)d_in[13];
    const float* W_vt = (const float*)d_in[14]; const float* b_vt = (const float*)d_in[15];
    const float* W_o  = (const float*)d_in[16]; const float* b_o  = (const float*)d_in[17];
    const float* W_f  = (const float*)d_in[18]; const float* b_f  = (const float*)d_in[19];
    const float* nw   = (const float*)d_in[20]; const float* nb   = (const float*)d_in[21];
    const float* gat  = (const float*)d_in[22];
    const int*   eid  = (const int*)d_in[23];

    char* ws = (char*)d_ws;
    const size_t BIG  = 16384ull * 1024 * 4;   // 64 MiB fp32
    const size_t BIGH = BIG / 2;               // 32 MiB bf16
    const size_t SML  = 4096ull * 1024 * 4;    // 16 MiB fp32
    const size_t SMLH = SML / 2;               // 8 MiB bf16
    const size_t WH   = 1024ull * 1024 * 2;    // 2 MiB bf16 weight

    size_t off = 0;
    float* qa_f = (float*)(ws + off); off += BIG;
    float* qt_f = (float*)(ws + off); off += BIG;
    char*  kvb  = ws + off;                    // 4*SML fp32 k/v block
    float* ka_f = (float*)(kvb);
    float* va_f = (float*)(kvb + SML);
    float* kt_f = (float*)(kvb + 2 * SML);
    float* vt_f = (float*)(kvb + 3 * SML);     off += BIG;
    unsigned short* xb    = (unsigned short*)(ws + off); off += BIGH;
    unsigned short* hadpb = (unsigned short*)(ws + off); off += SMLH;
    unsigned short* htb   = (unsigned short*)(ws + off); off += SMLH;
    unsigned short* qa_b  = (unsigned short*)(ws + off); off += BIGH;
    unsigned short* qt_b  = (unsigned short*)(ws + off); off += BIGH;
    unsigned short* ka_b  = (unsigned short*)(ws + off); off += SMLH;
    unsigned short* kt_b  = (unsigned short*)(ws + off); off += SMLH;
    unsigned short* vaT   = (unsigned short*)(ws + off); off += SMLH;
    unsigned short* vtT   = (unsigned short*)(ws + off); off += SMLH;
    unsigned short* wb[8];
    for (int i = 0; i < 8; ++i) { wb[i] = (unsigned short*)(ws + off); off += WH; }

    unsigned short* attn_ob = (unsigned short*)kvb;           // k/v fp32 dead after convert
    unsigned short* xn_b    = (unsigned short*)(kvb + 2 * SML);
    float* o_out = qa_f;                                      // q_a fp32 dead after rope

    // bf16 operand preparation
    pack_adapter_bf16_kernel<<<2048, 256, 0, stream>>>(h_a, p, hadpb);
    cvt_bf16_kernel<<<16384, 256, 0, stream>>>(x,   xb);
    cvt_bf16_kernel<<<4096,  256, 0, stream>>>(h_t, htb);
    const float* Wsrc[8] = { W_qa, W_ka, W_va, W_qt, W_kt, W_vt, W_o, W_f };
    for (int i = 0; i < 8; ++i)
        cvt_w_bf16_kernel<<<1024, 256, 0, stream>>>(Wsrc[i], eid, wb[i]);

    dim3 gBig(8, 128), gSml(8, 32);
    gemm_bf16_async_kernel<<<gBig, 256, 0, stream>>>(xb,    wb[0], b_qa, eid, qa_f, 0);
    gemm_bf16_async_kernel<<<gBig, 256, 0, stream>>>(xb,    wb[3], b_qt, eid, qt_f, 0);
    gemm_bf16_async_kernel<<<gSml, 256, 0, stream>>>(hadpb, wb[1], b_ka, eid, ka_f, 0);
    gemm_bf16_async_kernel<<<gSml, 256, 0, stream>>>(hadpb, wb[2], b_va, eid, va_f, 0);
    gemm_bf16_async_kernel<<<gSml, 256, 0, stream>>>(htb,   wb[4], b_kt, eid, kt_f, 0);
    gemm_bf16_async_kernel<<<gSml, 256, 0, stream>>>(htb,   wb[5], b_vt, eid, vt_f, 0);

    rope_cvt_kernel<<<(16384 * 512) / 256, 256, 0, stream>>>(qa_f, qa_b, 2048);
    rope_cvt_kernel<<<(16384 * 512) / 256, 256, 0, stream>>>(qt_f, qt_b, 2048);
    rope_cvt_kernel<<<(4096 * 512) / 256, 256, 0, stream>>>(ka_f, ka_b, 512);
    rope_cvt_kernel<<<(4096 * 512) / 256, 256, 0, stream>>>(kt_f, kt_b, 512);
    cvtT_v_kernel<<<16384, 256, 0, stream>>>(va_f, vaT);
    cvtT_v_kernel<<<16384, 256, 0, stream>>>(vt_f, vtT);

    attn_kernel<<<dim3(64, 32), 128, 0, stream>>>(qa_b, qt_b, ka_b, kt_b, vaT, vtT, gat, eid, attn_ob);

    gemm_bf16_async_kernel<<<gBig, 256, 0, stream>>>(attn_ob, wb[6], b_o, eid, o_out, 0);
    add_ln_kernel<<<16384, 256, 0, stream>>>(o_out, x, nw, nb, eid, xn_b);
    gemm_bf16_async_kernel<<<gBig, 256, 0, stream>>>(xn_b, wb[7], b_f, eid, (float*)d_out, 1);
}
